// TransformerBlock_9680856285641
// MI455X (gfx1250) — matrix operand order repaired and checked by reading, not compiled, not run on hardware
//
#include <hip/hip_runtime.h>
#include <hip/hip_bf16.h>

// ---------------------------------------------------------------------------
// Transformer block for MI455X (gfx1250, wave32, WMMA).
// All GEMMs run on v_wmma_f32_16x16x32_bf16 (bf16 operands, f32 accumulate).
// Weights are pre-transposed to [N][K] bf16 so BOTH GEMM operand tiles stage
// via GLOBAL_LOAD_ASYNC_TO_LDS_B128 (plain 2D copies, no in-loop transpose).
// V is produced transposed so flash attention reads P*V fragments from L2.
// ---------------------------------------------------------------------------

typedef __attribute__((ext_vector_type(16))) __bf16 v16bf;
typedef __attribute__((ext_vector_type(8)))  __bf16 bf16x8;
typedef __attribute__((ext_vector_type(8)))  float  v8f;

#define D_MODEL 1024
#define N_HEADS 16
#define D_HEAD  64
#define D_MLP   4096
#define BATCH   2
#define SEQ     2048
#define ROWS    (BATCH * SEQ)   /* 4096 */

#if defined(__has_builtin)
#if __has_builtin(__builtin_amdgcn_global_load_async_to_lds_b128) && \
    __has_builtin(__builtin_amdgcn_s_wait_asynccnt)
#define USE_ASYNC_LDS 1
#endif
#endif
#ifndef USE_ASYNC_LDS
#define USE_ASYNC_LDS 0
#endif

#if USE_ASYNC_LDS
typedef __attribute__((ext_vector_type(4))) int i32x4;
typedef __attribute__((address_space(1))) i32x4 gbl_i32x4;
typedef __attribute__((address_space(3))) i32x4 lds_i32x4;
#endif

// ------------------------------- utilities --------------------------------

// dst[n*K + k] = (bf16) src[k*N + n]   (weight [K][N] f32 -> [N][K] bf16)
__global__ void transpose_to_bf16_kernel(const float* __restrict__ src,
                                         __bf16* __restrict__ dst, int K, int N) {
    int i = blockIdx.x * blockDim.x + threadIdx.x;   // over N*K (exact)
    int n = i / K;
    int k = i - n * K;
    dst[i] = (__bf16)src[(size_t)k * N + n];
}

// W_{Q,K,V}: [h, d, kq] f32 -> transposed [n=h*64+kq][d] bf16
__global__ void rearrange_qkv_wT_kernel(const float* __restrict__ src,
                                        __bf16* __restrict__ dst) {
    int i = blockIdx.x * blockDim.x + threadIdx.x;   // 0 .. 16*64*1024-1
    int n = i >> 10;          // h*64 + kq
    int d = i & 1023;
    int h = n >> 6, kq = n & 63;
    dst[i] = (__bf16)src[((size_t)h << 16) + d * 64 + kq];
}

// LayerNorm over D_MODEL=1024, one row per block (256 threads, 4 elems each).
__global__ __launch_bounds__(256)
void layernorm_kernel(const float* __restrict__ x, const float* __restrict__ w,
                      const float* __restrict__ b, __bf16* __restrict__ y) {
    __shared__ float s_sum[256];
    __shared__ float s_sq[256];
    const int tid = threadIdx.x;
    const int row = blockIdx.x;
    const float* xr = x + (size_t)row * D_MODEL;

    float v[4], sum = 0.f, sq = 0.f;
#pragma unroll
    for (int i = 0; i < 4; i++) {
        v[i] = xr[tid + i * 256];
        sum += v[i];
        sq  += v[i] * v[i];
    }
    s_sum[tid] = sum; s_sq[tid] = sq;
    __syncthreads();
    for (int s = 128; s > 0; s >>= 1) {
        if (tid < s) { s_sum[tid] += s_sum[tid + s]; s_sq[tid] += s_sq[tid + s]; }
        __syncthreads();
    }
    const float mean = s_sum[0] * (1.0f / D_MODEL);
    const float var  = s_sq[0] * (1.0f / D_MODEL) - mean * mean;
    const float inv  = rsqrtf(var + 1e-5f);
#pragma unroll
    for (int i = 0; i < 4; i++) {
        int c = tid + i * 256;
        y[(size_t)row * D_MODEL + c] = (__bf16)((v[i] - mean) * inv * w[c] + b[c]);
    }
}

// ------------------------------ WMMA GEMM ---------------------------------
// C[M,N] = A[M,K](bf16 row-major) * BT[N,K](bf16, pre-transposed weights)
// (+bias)(+relu)(+residual).  Outputs: f32 outF, bf16 outB (row-major), or
// bf16 outBT (transposed [N][ldcT], used for V so attention reads V^T).
// Block: 256 threads = 8 waves, tile 128x128, K-step 32, double-buffered LDS.
// Wave (wm, wn): 32x64 subtile = 2x4 WMMA fragments -> 8 WMMA per K-step.
__global__ __launch_bounds__(256)
void gemm_bf16_kernel(const __bf16* __restrict__ A, int lda,
                      const __bf16* __restrict__ BT, int ldb,
                      const float* __restrict__ bias,
                      const float* __restrict__ residual,
                      float* __restrict__ outF, __bf16* __restrict__ outB,
                      __bf16* __restrict__ outBT, int ldcT,
                      int ldc, int K, int relu) {
    __shared__ __align__(16) __bf16 sA[2][128 * 32];    // [row][k]
    __shared__ __align__(16) __bf16 sBT[2][128 * 32];   // [col][k]

    const int tid  = threadIdx.x;
    const int lane = tid & 31;
    const int wave = tid >> 5;
    const int l15  = lane & 15;
    const int half = lane >> 4;
    const int rowBase = blockIdx.y * 128;
    const int colBase = blockIdx.x * 128;
    const int wm = (wave & 3) * 32;
    const int wn = (wave >> 2) * 64;

    // Staging: 512 chunks of 8 bf16 per tile; chunk c -> line c>>2, kchunk c&3.
#if USE_ASYNC_LDS
    auto stageTiles = [&](int k0, int buf) {   // both operands: async DMA to LDS
#pragma unroll
        for (int i = 0; i < 2; i++) {
            int c = tid + i * 256;
            const __bf16* g = &A[(size_t)(rowBase + (c >> 2)) * lda + k0 + (c & 3) * 8];
            __builtin_amdgcn_global_load_async_to_lds_b128(
                (gbl_i32x4*)g, (lds_i32x4*)&sA[buf][(c >> 2) * 32 + (c & 3) * 8], 0, 0);
        }
#pragma unroll
        for (int i = 0; i < 2; i++) {
            int c = tid + i * 256;
            const __bf16* g = &BT[(size_t)(colBase + (c >> 2)) * ldb + k0 + (c & 3) * 8];
            __builtin_amdgcn_global_load_async_to_lds_b128(
                (gbl_i32x4*)g, (lds_i32x4*)&sBT[buf][(c >> 2) * 32 + (c & 3) * 8], 0, 0);
        }
    };
#else
    bf16x8 aReg[2], bReg[2];
    auto loadTiles = [&](int k0) {
#pragma unroll
        for (int i = 0; i < 2; i++) {
            int c = tid + i * 256;
            aReg[i] = *(const bf16x8*)&A[(size_t)(rowBase + (c >> 2)) * lda + k0 + (c & 3) * 8];
            bReg[i] = *(const bf16x8*)&BT[(size_t)(colBase + (c >> 2)) * ldb + k0 + (c & 3) * 8];
        }
    };
    auto storeTiles = [&](int buf) {
#pragma unroll
        for (int i = 0; i < 2; i++) {
            int c = tid + i * 256;
            *(bf16x8*)&sA[buf][(c >> 2) * 32 + (c & 3) * 8]  = aReg[i];
            *(bf16x8*)&sBT[buf][(c >> 2) * 32 + (c & 3) * 8] = bReg[i];
        }
    };
#endif

    v8f acc[2][4] = {};
    const int nT = K / 32;

#if USE_ASYNC_LDS
    stageTiles(0, 0);
#else
    loadTiles(0);
    storeTiles(0);
#endif

    for (int t = 0; t < nT; t++) {
        const int buf = t & 1;
#if USE_ASYNC_LDS
        __builtin_amdgcn_s_wait_asynccnt(0);   // drain this wave's async copies
#endif
        __syncthreads();
        if (t + 1 < nT) {
#if USE_ASYNC_LDS
            stageTiles((t + 1) * 32, buf ^ 1); // DMA next tiles into idle buffer
#else
            loadTiles((t + 1) * 32);
#endif
            if (t + 2 < nT) {   // pull tile t+2 toward L2 early (global_prefetch_b8)
                __builtin_prefetch(&A[(size_t)(rowBase + (tid >> 2)) * lda + (t + 2) * 32], 0, 1);
                __builtin_prefetch(&BT[(size_t)(colBase + (tid >> 2)) * ldb + (t + 2) * 32], 0, 1);
            }
        }

        // fragments
        v16bf afrag[2], bfrag[4];
#pragma unroll
        for (int i = 0; i < 2; i++) {
            int row = wm + i * 16 + l15;
            int d0  = half ? 8 : 0;    // A layout: lo half k{0..7,16..23}, hi {8..15,24..31}
            bf16x8 lo = *(const bf16x8*)&sA[buf][row * 32 + d0];
            bf16x8 hi = *(const bf16x8*)&sA[buf][row * 32 + d0 + 16];
#pragma unroll
            for (int e = 0; e < 8; e++) { afrag[i][e] = lo[e]; afrag[i][e + 8] = hi[e]; }
        }
#pragma unroll
        for (int j = 0; j < 4; j++) {
            int col = wn + j * 16 + l15;
            int k0b = half ? 8 : 0;    // B layout: lo half k{0..7,16..23}, hi {8..15,24..31}
            bf16x8 lo = *(const bf16x8*)&sBT[buf][col * 32 + k0b];
            bf16x8 hi = *(const bf16x8*)&sBT[buf][col * 32 + k0b + 16];
#pragma unroll
            for (int e = 0; e < 8; e++) { bfrag[j][e] = lo[e]; bfrag[j][e + 8] = hi[e]; }
        }
#pragma unroll
        for (int i = 0; i < 2; i++)
#pragma unroll
            for (int j = 0; j < 4; j++)
                acc[i][j] = __builtin_amdgcn_wmma_f32_16x16x32_bf16(
                    false, afrag[i], false, bfrag[j], (short)0, acc[i][j], false, false);

#if !USE_ASYNC_LDS
        if (t + 1 < nT) storeTiles(buf ^ 1);   // other buffer: read at t+1 after barrier
#endif
    }

    // --- epilogue: C layout row = r + 8*half, col = l15 ---
#pragma unroll
    for (int i = 0; i < 2; i++)
#pragma unroll
        for (int j = 0; j < 4; j++) {
            int col  = colBase + wn + j * 16 + l15;
            int row0 = rowBase + wm + i * 16 + half * 8;
            bf16x8 tpack;
#pragma unroll
            for (int r = 0; r < 8; r++) {
                int row = row0 + r;
                float vv = acc[i][j][r];
                if (bias) vv += bias[col];
                if (relu) vv = fmaxf(vv, 0.f);
                size_t idx = (size_t)row * ldc + col;
                if (residual) vv += residual[idx];
                if (outF) outF[idx] = vv;
                if (outB) outB[idx] = (__bf16)vv;
                tpack[r] = (__bf16)vv;
            }
            if (outBT)   // transposed bf16 output: 8 consecutive rows per lane
                *(bf16x8*)&outBT[(size_t)col * ldcT + row0] = tpack;
        }
}

// ---------------------------- flash attention -----------------------------
// One wave per (batch, head, 32 query rows). Keys streamed in tiles of 32,
// causal, online softmax. Q/K stored bf16 [B*S, H*64]; V stored transposed
// bf16 [H*64, B*S] so P*V B-fragments are contiguous global reads (L2-hot).
// 16 WMMA per key tile (8 for Q*K^T over 2 row-frags, 8 for P*V).
__global__ __launch_bounds__(32)
void attention_kernel(const __bf16* __restrict__ Q, const __bf16* __restrict__ Km,
                      const __bf16* __restrict__ VT, __bf16* __restrict__ Z) {
    __shared__ __align__(16) __bf16 sP[32 * 32];    // probs [qrow][key]

    const int lane = threadIdx.x;
    const int l15  = lane & 15;
    const int half = lane >> 4;
    const int qBase = blockIdx.x * 32;
    const int h = blockIdx.y, b = blockIdx.z;
    const size_t rs = D_MODEL;
    const __bf16* Qb  = Q  + ((size_t)b * SEQ) * rs + h * D_HEAD;
    const __bf16* Kb  = Km + ((size_t)b * SEQ) * rs + h * D_HEAD;
    const __bf16* VTb = VT + (size_t)h * D_HEAD * ROWS + (size_t)b * SEQ;  // [d][key]

    // Q fragments: [row-frag i][d-chunk kc], A layout, loaded once.
    v16bf qf[2][2];
#pragma unroll
    for (int i = 0; i < 2; i++)
#pragma unroll
        for (int f = 0; f < 2; f++) {
            const __bf16* qr = Qb + (size_t)(qBase + i * 16 + l15) * rs + f * 32;
            int d0 = half ? 8 : 0;
            bf16x8 lo = *(const bf16x8*)&qr[d0];
            bf16x8 hi = *(const bf16x8*)&qr[d0 + 16];
#pragma unroll
            for (int e = 0; e < 8; e++) { qf[i][f][e] = lo[e]; qf[i][f][e + 8] = hi[e]; }
        }

    v8f zacc[2][4] = {};
    float mrow[2][8], lrow[2][8];
#pragma unroll
    for (int i = 0; i < 2; i++)
#pragma unroll
        for (int r = 0; r < 8; r++) { mrow[i][r] = -1e30f; lrow[i][r] = 0.f; }
    const float scale = 0.125f;   // 1/sqrt(64)
    const int qEnd = qBase + 32;  // 32-aligned: kb+31 <= qEnd-1 <= SEQ-1, no clamps

    for (int kb = 0; kb < qEnd; kb += 32) {
        // scores: S = Q * K^T. K rows load directly as B fragments (d contiguous).
        v8f sf[2][2] = {};
#pragma unroll
        for (int sub = 0; sub < 2; sub++) {
            const __bf16* kr = Kb + (size_t)(kb + sub * 16 + l15) * rs;
#pragma unroll
            for (int kc = 0; kc < 2; kc++) {
                int d0 = kc * 32 + (half ? 8 : 0);
                bf16x8 lo = *(const bf16x8*)&kr[d0];
                bf16x8 hi = *(const bf16x8*)&kr[d0 + 16];
                v16bf bfr;
#pragma unroll
                for (int e = 0; e < 8; e++) { bfr[e] = lo[e]; bfr[e + 8] = hi[e]; }
#pragma unroll
                for (int i = 0; i < 2; i++)
                    sf[i][sub] = __builtin_amdgcn_wmma_f32_16x16x32_bf16(
                        false, qf[i][kc], false, bfr, (short)0, sf[i][sub], false, false);
            }
        }

        // online softmax per row-fragment (reductions within each 16-lane half)
#pragma unroll
        for (int i = 0; i < 2; i++) {
            float alpha[8];
#pragma unroll
            for (int r = 0; r < 8; r++) {
                int qrow = qBase + i * 16 + r + half * 8;
#pragma unroll
                for (int sub = 0; sub < 2; sub++) {
                    int key = kb + sub * 16 + l15;
                    float sv = sf[i][sub][r] * scale;
                    if (key > qrow) sv = -1e30f;   // causal mask
                    sf[i][sub][r] = sv;
                }
                float vmax = fmaxf(sf[i][0][r], sf[i][1][r]);
#pragma unroll
                for (int m = 1; m < 16; m <<= 1) vmax = fmaxf(vmax, __shfl_xor(vmax, m, 32));
                float mnew = fmaxf(mrow[i][r], vmax);
                alpha[r] = __expf(mrow[i][r] - mnew);
                float p0 = __expf(sf[i][0][r] - mnew);
                float p1 = __expf(sf[i][1][r] - mnew);
                sf[i][0][r] = p0; sf[i][1][r] = p1;
                float ps = p0 + p1;
#pragma unroll
                for (int m = 1; m < 16; m <<= 1) ps += __shfl_xor(ps, m, 32);
                lrow[i][r] = lrow[i][r] * alpha[r] + ps;
                mrow[i][r] = mnew;
            }
#pragma unroll
            for (int nd = 0; nd < 4; nd++)
#pragma unroll
                for (int r = 0; r < 8; r++) zacc[i][nd][r] *= alpha[r];
            // C-layout probs -> LDS (bf16) for A-layout reload
#pragma unroll
            for (int sub = 0; sub < 2; sub++)
#pragma unroll
                for (int r = 0; r < 8; r++)
                    sP[(i * 16 + r + half * 8) * 32 + sub * 16 + l15] = (__bf16)sf[i][sub][r];
        }
        __syncthreads();

        v16bf pf[2];
#pragma unroll
        for (int i = 0; i < 2; i++) {
            int d0 = half ? 8 : 0;
            bf16x8 lo = *(const bf16x8*)&sP[(i * 16 + l15) * 32 + d0];
            bf16x8 hi = *(const bf16x8*)&sP[(i * 16 + l15) * 32 + d0 + 16];
#pragma unroll
            for (int e = 0; e < 8; e++) { pf[i][e] = lo[e]; pf[i][e + 8] = hi[e]; }
        }
        // Z += P * V: V^T fragments straight from global (contiguous keys)
#pragma unroll
        for (int nd = 0; nd < 4; nd++) {
            const __bf16* vr = VTb + (size_t)(nd * 16 + l15) * ROWS + kb + (half ? 8 : 0);
            bf16x8 lo = *(const bf16x8*)&vr[0];
            bf16x8 hi = *(const bf16x8*)&vr[16];
            v16bf vf;
#pragma unroll
            for (int e = 0; e < 8; e++) { vf[e] = lo[e]; vf[e + 8] = hi[e]; }
#pragma unroll
            for (int i = 0; i < 2; i++)
                zacc[i][nd] = __builtin_amdgcn_wmma_f32_16x16x32_bf16(
                    false, pf[i], false, vf, (short)0, zacc[i][nd], false, false);
        }
        __syncthreads();
    }

    // finalize: divide by softmax denominator, write bf16 Z[B*S, H*64]
    __bf16* Zb = Z + ((size_t)b * SEQ) * rs + h * D_HEAD;
#pragma unroll
    for (int i = 0; i < 2; i++)
#pragma unroll
        for (int nd = 0; nd < 4; nd++)
#pragma unroll
            for (int r = 0; r < 8; r++) {
                float vv = zacc[i][nd][r] / lrow[i][r];
                Zb[(size_t)(qBase + i * 16 + r + half * 8) * rs + nd * 16 + l15] = (__bf16)vv;
            }
}

// ------------------------------- launcher ---------------------------------

extern "C" void kernel_launch(void* const* d_in, const int* in_sizes, int n_in,
                              void* d_out, int out_size, void* d_ws, size_t ws_size,
                              hipStream_t stream) {
    const float* resid_pre = (const float*)d_in[0];
    const float* ln1_w = (const float*)d_in[1];
    const float* ln1_b = (const float*)d_in[2];
    const float* W_Q   = (const float*)d_in[3];
    const float* b_Q   = (const float*)d_in[4];
    const float* W_K   = (const float*)d_in[5];
    const float* b_K   = (const float*)d_in[6];
    const float* W_V   = (const float*)d_in[7];
    const float* b_V   = (const float*)d_in[8];
    const float* W_O   = (const float*)d_in[9];
    const float* b_O   = (const float*)d_in[10];
    const float* ln2_w = (const float*)d_in[11];
    const float* ln2_b = (const float*)d_in[12];
    const float* W_in  = (const float*)d_in[13];
    const float* b_in  = (const float*)d_in[14];
    const float* W_out = (const float*)d_in[15];
    const float* b_out = (const float*)d_in[16];

    char* ws = (char*)d_ws;
    size_t off = 0;
    auto alloc = [&](size_t bytes) -> void* {
        void* p = ws + off;
        off += (bytes + 255) & ~(size_t)255;
        return p;
    };
    const size_t nQKV = (size_t)N_HEADS * D_MODEL * D_HEAD;   // 1,048,576
    __bf16* wqT   = (__bf16*)alloc(nQKV * 2);                 // [N=1024][K=1024]
    __bf16* wkT   = (__bf16*)alloc(nQKV * 2);
    __bf16* wvT   = (__bf16*)alloc(nQKV * 2);
    __bf16* woT   = (__bf16*)alloc((size_t)D_MODEL * D_MODEL * 2);
    __bf16* winT  = (__bf16*)alloc((size_t)D_MODEL * D_MLP * 2);   // [4096][1024]
    __bf16* woutT = (__bf16*)alloc((size_t)D_MLP * D_MODEL * 2);   // [1024][4096]
    __bf16* xln   = (__bf16*)alloc((size_t)ROWS * D_MODEL * 2);
    __bf16* q_bf  = (__bf16*)alloc((size_t)ROWS * D_MODEL * 2);
    __bf16* k_bf  = (__bf16*)alloc((size_t)ROWS * D_MODEL * 2);
    __bf16* vT_bf = (__bf16*)alloc((size_t)D_MODEL * ROWS * 2);    // [H*64][B*S]
    __bf16* z_bf  = (__bf16*)alloc((size_t)ROWS * D_MODEL * 2);
    float*  resid_mid = (float*)alloc((size_t)ROWS * D_MODEL * 4);
    __bf16* hln   = (__bf16*)alloc((size_t)ROWS * D_MODEL * 2);
    __bf16* hidden= (__bf16*)alloc((size_t)ROWS * D_MLP * 2);
    (void)ws_size;

    // weight conversion: all weights -> transposed [N][K] bf16
    rearrange_qkv_wT_kernel<<<(unsigned)(nQKV / 256), 256, 0, stream>>>(W_Q, wqT);
    rearrange_qkv_wT_kernel<<<(unsigned)(nQKV / 256), 256, 0, stream>>>(W_K, wkT);
    rearrange_qkv_wT_kernel<<<(unsigned)(nQKV / 256), 256, 0, stream>>>(W_V, wvT);
    transpose_to_bf16_kernel<<<(unsigned)(D_MODEL * D_MODEL / 256), 256, 0, stream>>>(
        W_O, woT, D_MODEL, D_MODEL);
    transpose_to_bf16_kernel<<<(unsigned)(D_MODEL * D_MLP / 256), 256, 0, stream>>>(
        W_in, winT, D_MODEL, D_MLP);     // K=1024, N=4096
    transpose_to_bf16_kernel<<<(unsigned)(D_MLP * D_MODEL / 256), 256, 0, stream>>>(
        W_out, woutT, D_MLP, D_MODEL);   // K=4096, N=1024

    // LN1
    layernorm_kernel<<<ROWS, 256, 0, stream>>>(resid_pre, ln1_w, ln1_b, xln);

    // QKV projections (N = H*64 = 1024; V written transposed for attention)
    dim3 gProj(D_MODEL / 128, ROWS / 128);
    gemm_bf16_kernel<<<gProj, 256, 0, stream>>>(xln, D_MODEL, wqT, D_MODEL,
        b_Q, nullptr, nullptr, q_bf, nullptr, 0, D_MODEL, D_MODEL, 0);
    gemm_bf16_kernel<<<gProj, 256, 0, stream>>>(xln, D_MODEL, wkT, D_MODEL,
        b_K, nullptr, nullptr, k_bf, nullptr, 0, D_MODEL, D_MODEL, 0);
    gemm_bf16_kernel<<<gProj, 256, 0, stream>>>(xln, D_MODEL, wvT, D_MODEL,
        b_V, nullptr, nullptr, nullptr, vT_bf, ROWS, D_MODEL, D_MODEL, 0);

    // attention (32 query rows per wave)
    attention_kernel<<<dim3(SEQ / 32, N_HEADS, BATCH), 32, 0, stream>>>(q_bf, k_bf, vT_bf, z_bf);

    // output projection + bias + residual -> resid_mid (f32)
    gemm_bf16_kernel<<<gProj, 256, 0, stream>>>(z_bf, D_MODEL, woT, D_MODEL,
        b_O, resid_pre, resid_mid, nullptr, nullptr, 0, D_MODEL, D_MODEL, 0);

    // LN2
    layernorm_kernel<<<ROWS, 256, 0, stream>>>(resid_mid, ln2_w, ln2_b, hln);

    // MLP up + ReLU -> bf16 hidden
    gemm_bf16_kernel<<<dim3(D_MLP / 128, ROWS / 128), 256, 0, stream>>>(
        hln, D_MODEL, winT, D_MODEL,
        b_in, nullptr, nullptr, hidden, nullptr, 0, D_MLP, D_MODEL, 1);

    // MLP down + bias + residual -> d_out (f32)
    gemm_bf16_kernel<<<gProj, 256, 0, stream>>>(hidden, D_MLP, woutT, D_MLP,
        b_out, resid_mid, (float*)d_out, nullptr, nullptr, 0, D_MODEL, D_MLP, 0);
}
